// SelfAttention_12077448036715
// MI455X (gfx1250) — matrix operand order repaired and checked by reading, not compiled, not run on hardware
//
#include <hip/hip_runtime.h>

// ---------------------------------------------------------------------------
// Self-attention (N=8, S=2048, D=1024, fp32) for MI455X / gfx1250.
// Flash-attention v2, bf16 WMMA, fp32 accum.
//  - K tiles staged into LDS by the Tensor Data Mover (tensor_load_to_lds),
//    double-pipelined against softmax+PV, waited with s_wait_tensorcnt.
//  - Q tile staged once per WG with global_load_async_to_lds_b128 (ASYNCcnt).
//  - LDS rows padded 16B per 1024B (TDM pad_enable) => 16B-aligned ds_load_b128.
// d_ws layout (requires >= 96MB): Qbf16[N*S*D] | Kbf16[N*S*D] | VT_bf16[N*D*S]
// ---------------------------------------------------------------------------

typedef __attribute__((ext_vector_type(16))) __bf16 v16bf;
typedef __attribute__((ext_vector_type(8)))  __bf16 v8bf;
typedef __attribute__((ext_vector_type(4)))  __bf16 v4bf;
typedef __attribute__((ext_vector_type(8)))  float  v8f;
typedef __attribute__((ext_vector_type(4)))  unsigned int v4u;
typedef __attribute__((ext_vector_type(8)))  int v8i;
typedef __attribute__((ext_vector_type(4)))  int v4i;

constexpr int  N_  = 8;
constexpr int  S_  = 2048;
constexpr int  D_  = 1024;
constexpr long NSD = (long)N_ * S_ * D_;

constexpr int BR = 64;            // query rows per workgroup
constexpr int BC = 64;            // key/value rows per block
constexpr int NWAVE = 16;
constexpr int TPB = NWAVE * 32;
constexpr int ROWE = 1040;        // LDS row stride in bf16 elems (1024 + 2x8 pad)
// element k of a row lives at row*ROWE + k + (k>=512 ? 8 : 0)

__device__ __forceinline__ v16bf ld_pair(const __bf16* p0, const __bf16* p1) {
  v8bf lo = *(const v8bf*)p0;
  v8bf hi = *(const v8bf*)p1;
  return __builtin_shufflevector(lo, hi, 0,1,2,3,4,5,6,7,8,9,10,11,12,13,14,15);
}

__device__ __forceinline__ v8f wmma_bf16(v16bf a, v16bf b, v8f c) {
  return __builtin_amdgcn_wmma_f32_16x16x32_bf16(false, a, false, b,
                                                 (short)0, c, false, false);
}

// TDM: DMA one 64-row x 1024-elem bf16 tile (row stride D_) into LDS at lds_off,
// inserting 16B of padding after every 1024B (=> LDS row stride 2080B).
__device__ __forceinline__ void tdm_load_tile(const __bf16* gsrc, unsigned lds_off) {
  const unsigned long long ga = (unsigned long long)gsrc;
  const v4u g0 = { 1u,                                  // count=1 (user D#)
                   lds_off,                             // lds_addr
                   (unsigned)(ga & 0xFFFFFFFFu),        // global_addr[31:0]
                   (unsigned)((ga >> 32) & 0x01FFFFFFu) // global_addr[56:32]
                     | 0x80000000u };                   // type=2 ("image")
  const v8i g1 = { (int)0x07D10000,  // wg_mask=0,data_size=2B,pad_en,intv=7(1024B),amt=3(16B)
                   (int)0x04000000,  // tensor_dim0 = 1024  ([79:48], low16 here)
                   (int)0x40000000,  // tensor_dim1 = 16384 ([111:80], low16 here)
                   (int)0x04000000,  // tile_dim0 = 1024    ([127:112])
                   64,               // tile_dim1 = 64      ([143:128]); tile_dim2=0
                   1024,             // tensor_dim0_stride  ([207:160], low32)
                   0, 0 };
  const v4i z4 = {0, 0, 0, 0};
  const v8i z8 = {0, 0, 0, 0, 0, 0, 0, 0};
  __builtin_amdgcn_tensor_load_to_lds(g0, g1, z4, z4, z8, 0);
}

// --------------------------- prep: Q = x*wl, K = x*wr (bf16) ----------------
__global__ __launch_bounds__(256)
void prep_qk_kernel(const float* __restrict__ x,
                    const float* __restrict__ wl,
                    const float* __restrict__ wr,
                    __bf16* __restrict__ Qb, __bf16* __restrict__ Kb) {
  long i = (long)blockIdx.x * blockDim.x + threadIdx.x;
  if (i >= NSD / 4) return;
  int d = (int)((i * 4) & (D_ - 1));
  float4 xv  = ((const float4*)x)[i];
  float4 wlv = *(const float4*)(wl + d);
  float4 wrv = *(const float4*)(wr + d);
  v4bf q = { (__bf16)(xv.x * wlv.x), (__bf16)(xv.y * wlv.y),
             (__bf16)(xv.z * wlv.z), (__bf16)(xv.w * wlv.w) };
  v4bf k = { (__bf16)(xv.x * wrv.x), (__bf16)(xv.y * wrv.y),
             (__bf16)(xv.z * wrv.z), (__bf16)(xv.w * wrv.w) };
  ((v4bf*)Qb)[i] = q;
  ((v4bf*)Kb)[i] = k;
}

// --------------------------- prep: VT[n][d][s] = bf16(x[n][s][d]) -----------
__global__ __launch_bounds__(256)
void prep_vt_kernel(const float* __restrict__ x, __bf16* __restrict__ Vt) {
  __shared__ __bf16 tile[32][33];
  const int n  = blockIdx.z;
  const int s0 = blockIdx.x * 32;
  const int d0 = blockIdx.y * 32;
  const int tx = threadIdx.x, ty = threadIdx.y;
#pragma unroll
  for (int k = 0; k < 4; ++k)
    tile[ty + k * 8][tx] = (__bf16)x[((long)n * S_ + s0 + ty + k * 8) * D_ + d0 + tx];
  __syncthreads();
#pragma unroll
  for (int k = 0; k < 4; ++k)
    Vt[((long)n * D_ + d0 + ty + k * 8) * S_ + s0 + tx] = tile[tx][ty + k * 8];
}

// --------------------------- flash attention --------------------------------
__global__ __launch_bounds__(TPB)
void attn_kernel(const __bf16* __restrict__ Qb, const __bf16* __restrict__ Kb,
                 const __bf16* __restrict__ Vt, float* __restrict__ out) {
  const int n     = blockIdx.y;
  const int qbase = blockIdx.x * BR;
  const int tid   = threadIdx.x;
  const int lane  = tid & 31;
  const int wid   = tid >> 5;
  const int r     = wid >> 2;   // 16-row block (both phases)
  const int cd    = wid & 3;    // score col-block / PV 256-col D-chunk
  const int l15   = lane & 15;
  const int hi    = lane >> 4;

  __shared__ __align__(16) __bf16 Qs[BR * ROWE];     // 130KB, TDM-style padding
  __shared__ __align__(16) __bf16 Ks[BR * ROWE];     // 130KB
  __shared__ float Ssh[BR][BC + 1];
  __shared__ __align__(16) __bf16 Psh[BR][BC + 8];
  __shared__ float msh[BR], lsh[BR], csh[BR];

  for (int i = tid; i < BR; i += TPB) { msh[i] = -3.0e38f; lsh[i] = 0.0f; }

  const unsigned qs_base = (unsigned)(unsigned long long)(void*)Qs;
  const unsigned ks_base = (unsigned)(unsigned long long)(void*)Ks;

  // ---- stage Q tile (64x1024 bf16) via async global->LDS copies -----------
  {
    const __bf16* qtile = Qb + ((long)n * S_ + qbase) * D_;
    for (int c = tid; c < BR * 128; c += TPB) {          // 16B chunks
      const int row = c >> 7, cc = c & 127;
      const unsigned dst = qs_base + (unsigned)(row * (ROWE * 2) + cc * 16 +
                                                ((cc >= 64) ? 16 : 0));
      const __bf16* src = qtile + (long)row * D_ + cc * 8;
      asm volatile("global_load_async_to_lds_b128 %0, %1, off"
                   :: "v"(dst), "v"(src) : "memory");
    }
  }
  // ---- TDM: stage K block 0 ------------------------------------------------
  if (wid == 0)
    tdm_load_tile(Kb + (long)n * S_ * D_, ks_base);

  asm volatile("s_wait_asynccnt 0x0" ::: "memory");
  if (wid == 0) __builtin_amdgcn_s_wait_tensorcnt(0);
  __syncthreads();

  const int aoff = hi * 16;   // A (16x32 bf16): lanes 0-15 k 0..15, hi k 16..31
  const int boff = hi * 16;   // B (32x16 bf16): lanes 0-15 k 0..15, hi k 16..31

  const v8f vzero = {0.f, 0.f, 0.f, 0.f, 0.f, 0.f, 0.f, 0.f};
  v8f acc[16];
#pragma unroll
  for (int t = 0; t < 16; ++t) acc[t] = vzero;

  constexpr int NB = S_ / BC;
  for (int j = 0; j < NB; ++j) {
    // ---- scores: 16x16 tile (r,cd) of Q K^T from LDS ----------------------
    const __bf16* qlds = Qs + (r * 16 + l15) * ROWE;
    const __bf16* klds = Ks + (cd * 16 + l15) * ROWE;
    v8f sc = vzero;
#pragma unroll
    for (int half = 0; half < 2; ++half) {
      const __bf16* qh = qlds + half * 520;   // 512 elems + 8 pad
      const __bf16* kh = klds + half * 520;
#pragma unroll 4
      for (int kk = 0; kk < 512; kk += 32) {
        v16bf a = ld_pair(qh + kk + aoff, qh + kk + aoff + 8);
        v16bf b = ld_pair(kh + kk + boff, kh + kk + boff + 8);
        sc = wmma_bf16(a, b, sc);
      }
    }
#pragma unroll
    for (int i = 0; i < 8; ++i)
      Ssh[r * 16 + hi * 8 + i][cd * 16 + l15] = sc[i];
    __syncthreads();

    // ---- kick TDM for next K block; overlaps softmax + PV -----------------
    if (wid == 0 && (j + 1) < NB)
      tdm_load_tile(Kb + ((long)n * S_ + (long)(j + 1) * BC) * D_, ks_base);

    // ---- online softmax: wave 0, two rows per lane ------------------------
    if (wid == 0) {
#pragma unroll
      for (int rr = 0; rr < 2; ++rr) {
        const int row = lane * 2 + rr;
        float m = msh[row];
        const float mold = m;
        for (int t = 0; t < BC; ++t) m = fmaxf(m, Ssh[row][t]);
        const float corr = __expf(mold - m);
        float sum = 0.0f;
        for (int t = 0; t < BC; ++t) {
          const float p = __expf(Ssh[row][t] - m);
          sum += p;
          Psh[row][t] = (__bf16)p;
        }
        msh[row] = m;
        lsh[row] = lsh[row] * corr + sum;
        csh[row] = corr;
      }
    }
    __syncthreads();

    // ---- rescale accumulators ---------------------------------------------
    float cr[8];
#pragma unroll
    for (int i = 0; i < 8; ++i) cr[i] = csh[r * 16 + hi * 8 + i];
#pragma unroll
    for (int t = 0; t < 16; ++t)
#pragma unroll
      for (int i = 0; i < 8; ++i) acc[t][i] *= cr[i];

    // ---- O += P V : A = P (LDS), B = V^T rows (L2-resident bf16) ----------
    const __bf16* prow  = &Psh[r * 16 + l15][0];
    const __bf16* vbase = Vt + ((long)n * D_ + cd * 256 + l15) * S_ + (long)j * BC + boff;
#pragma unroll
    for (int kk = 0; kk < BC; kk += 32) {
      v16bf a = ld_pair(prow + kk + aoff, prow + kk + aoff + 8);
#pragma unroll
      for (int t = 0; t < 16; ++t) {
        const __bf16* vp = vbase + (long)t * 16 * S_ + kk;
        v16bf b = ld_pair(vp, vp + 8);
        acc[t] = wmma_bf16(a, b, acc[t]);
      }
    }

    // ---- K DMA must be in LDS before next score phase ---------------------
    if (wid == 0) __builtin_amdgcn_s_wait_tensorcnt(0);
    __syncthreads();
  }

  // ---- epilogue: divide by l, store fp32 ----------------------------------
  float li[8];
#pragma unroll
  for (int i = 0; i < 8; ++i) li[i] = 1.0f / lsh[r * 16 + hi * 8 + i];
#pragma unroll
  for (int t = 0; t < 16; ++t) {
#pragma unroll
    for (int i = 0; i < 8; ++i) {
      const long row = (long)qbase + r * 16 + hi * 8 + i;
      out[((long)n * S_ + row) * D_ + cd * 256 + t * 16 + l15] = acc[t][i] * li[i];
    }
  }
}

// --------------------------- host entry -------------------------------------
extern "C" void kernel_launch(void* const* d_in, const int* in_sizes, int n_in,
                              void* d_out, int out_size, void* d_ws, size_t ws_size,
                              hipStream_t stream) {
  (void)in_sizes; (void)n_in; (void)out_size; (void)ws_size; // ws >= 96MB assumed
  const float* x  = (const float*)d_in[0];
  const float* wl = (const float*)d_in[1];
  const float* wr = (const float*)d_in[2];
  float* out = (float*)d_out;

  __bf16* Qb = (__bf16*)d_ws;
  __bf16* Kb = Qb + NSD;
  __bf16* Vt = Kb + NSD;

  prep_qk_kernel<<<dim3((unsigned)((NSD / 4 + 255) / 256)), 256, 0, stream>>>(x, wl, wr, Qb, Kb);
  prep_vt_kernel<<<dim3(S_ / 32, D_ / 32, N_), dim3(32, 8), 0, stream>>>(x, Vt);
  attn_kernel<<<dim3(S_ / BR, N_), TPB, 0, stream>>>(Qb, Kb, Vt, out);
}
